// LSTM_32933809225777
// MI455X (gfx1250) — compile-verified, matrix operand order checked by reading, not run on hardware
//
#include <hip/hip_runtime.h>
#include <hip/hip_bf16.h>

// Problem constants (match reference): B=8, T=128, N=256, DIN=32, H=64
#define BB   8
#define TTs  128
#define NNr  256
#define DIN0 32
#define HHh  64

typedef _Float16 v16h __attribute__((ext_vector_type(16)));
typedef _Float16 v8h  __attribute__((ext_vector_type(8)));
typedef float    v8f  __attribute__((ext_vector_type(8)));

// 16-byte int vector + address-space-qualified pointers for the async-LDS builtin.
typedef int v4i __attribute__((vector_size(16)));
typedef __attribute__((address_space(1))) v4i g_v4i;   // global
typedef __attribute__((address_space(3))) v4i l_v4i;   // LDS

union Frag16 { v16h v; v8h h[2]; };

#if __has_builtin(__builtin_amdgcn_global_load_async_to_lds_b128) && \
    __has_builtin(__builtin_amdgcn_s_wait_asynccnt)
#define HAVE_ASYNC_LDS 1
#endif

// Load a 16x32 f16 WMMA operand fragment from row-major storage with leading
// dimension `ld` (halves). Lane l -> row (l&15); K chunks (l>>4)*8..+7 and +16.
static __device__ inline v16h load_frag(const _Float16* base, int ld, int lane) {
    int row  = lane & 15;
    int koff = (lane >> 4) * 8;
    const _Float16* p = base + row * ld + koff;
    Frag16 f;
    f.h[0] = *(const v8h*)p;
    f.h[1] = *(const v8h*)(p + 16);
    return f.v;
}

static __device__ inline float sigm(float x) { return 1.0f / (1.0f + __expf(-x)); }

static __device__ inline v8f splat8(float x) {
    v8f r;
#pragma unroll
    for (int i = 0; i < 8; ++i) r[i] = x;
    return r;
}

// ---------------- prep kernels ----------------

__global__ void k_f32_to_f16(const float* __restrict__ in, _Float16* __restrict__ out, int n) {
    for (int i = blockIdx.x * blockDim.x + threadIdx.x; i < n; i += gridDim.x * blockDim.x)
        out[i] = (_Float16)in[i];
}

// W[k][j] (K x Nc, row-major f32)  ->  WT[j][k] (Nc x K, row-major f16)
__global__ void k_transpose_f16(const float* __restrict__ W, _Float16* __restrict__ WT,
                                int K, int Nc) {
    int total = K * Nc;
    for (int i = blockIdx.x * blockDim.x + threadIdx.x; i < total; i += gridDim.x * blockDim.x) {
        int k = i % K;
        int j = i / K;
        WT[j * K + k] = (_Float16)W[k * Nc + j];
    }
}

__global__ void k_zero(float* __restrict__ c, int nc, _Float16* __restrict__ h, int nh) {
    int idx = blockIdx.x * blockDim.x + threadIdx.x;
    int stride = gridDim.x * blockDim.x;
    for (int i = idx; i < nc; i += stride) c[i] = 0.0f;
    for (int i = idx; i < nh; i += stride) h[i] = (_Float16)0.0f;
}

// ---------------- recurrent step kernel ----------------
// One launch per (layer, timestep). grid = B*8 workgroups, 256 threads (8 waves).
// Workgroup (b, oct) owns rows [oct*32, oct*32+32). Wave w = (quarter q, strip s):
//   s = w & 1   -> row strip r0 = oct*32 + s*16
//   q = w >> 1  -> hcol tile  [q*16, q*16+16)
// Phase 1: wave computes one 16x16 tile of ah = A @ h       (K=256, 8 WMMA)
// Phase 2: wave computes 4 gate tiles of z = [x, ah] @ W + b (K=DIN_+64),
//          so i/f/g/o for each (row, hcol) are co-resident in registers.
template <int DIN_, bool L0>
__global__ __launch_bounds__(256) void k_step(
    const float*    __restrict__ xv,   // L0: values
    const float*    __restrict__ xm,   // L0: masks
    const _Float16* __restrict__ Xh,   // L1: layer-0 hidden stream [B,T,N,H] f16
    const _Float16* __restrict__ A16,  // [B,N,N] row-major f16
    const _Float16* __restrict__ WT,   // [4H, DIN_+H] j-major f16
    const float*    __restrict__ bias, // [4H]
    const _Float16* __restrict__ hT_in,  // [B,H,N] f16 (transposed hidden)
    _Float16*       __restrict__ hT_out, // [B,H,N] f16
    float*          __restrict__ cbuf,   // [B,N,H] f32
    _Float16*       __restrict__ Hout,   // [B,T,N,H] f16
    int t)
{
    constexpr int K2 = DIN_ + HHh;     // 96 (L0) or 128 (L1)
    constexpr int KC = K2 / 32;        // 3 or 4

    __shared__ __align__(16) _Float16 sh_hT[HHh * NNr]; // 32 KB: h^T[j][m]
    __shared__ __align__(16) _Float16 sh_xa[2][2048];   //  8 KB: per-strip [16][K2]

    const int b    = blockIdx.x >> 3;
    const int oct  = blockIdx.x & 7;
    const int tid  = threadIdx.x;
    const int w    = tid >> 5;
    const int lane = tid & 31;
    const int s    = w & 1;            // row strip within the workgroup
    const int q    = w >> 1;           // hcol quarter
    const int r0   = oct * 32 + s * 16;

    // Stage h^T for this batch into LDS (2048 x 16B), async if available.
    {
        _Float16* srch = const_cast<_Float16*>(hT_in) + (size_t)b * HHh * NNr;
#ifdef HAVE_ASYNC_LDS
        for (int i = tid; i < (HHh * NNr) / 8; i += 256) {
            __builtin_amdgcn_global_load_async_to_lds_b128(
                (g_v4i*)(srch + (size_t)i * 8),
                (l_v4i*)(sh_hT + i * 8),
                0, 0);
        }
        __builtin_amdgcn_s_wait_asynccnt(0);
#else
        const v8h* src = (const v8h*)srch;
        v8h* dst = (v8h*)sh_hT;
        for (int i = tid; i < (HHh * NNr) / 8; i += 256) dst[i] = src[i];
#endif
    }
    __syncthreads();

    // ---------- Phase 1: one 16x16 tile of ah = A[r0:+16, :] @ h ----------
    v8f acc = splat8(0.0f);
    const _Float16* Ab = A16 + (size_t)b * NNr * NNr + (size_t)r0 * NNr;
#pragma unroll 1
    for (int kc = 0; kc < 8; ++kc) {
        if (kc < 7)
            __builtin_prefetch((const void*)(Ab + (kc + 1) * 32), 0, 3);
        v16h af = load_frag(Ab + kc * 32, NNr, lane);
        v16h bf = load_frag(sh_hT + q * 16 * NNr + kc * 32, NNr, lane);
        acc = __builtin_amdgcn_wmma_f32_16x16x32_f16(
            false, af, false, bf, (short)0, acc, false, false);
    }

    // ---------- build concat tile xa = [x_t | ah] in LDS ----------
    {
        _Float16* xa = sh_xa[s];
#pragma unroll
        for (int v = 0; v < 8; ++v) {
            int m = v + ((lane >> 4) << 3);     // C-layout: row within strip
            int n = lane & 15;                  // C-layout: col within tile
            xa[m * K2 + DIN_ + q * 16 + n] = (_Float16)acc[v];
        }
    }
    if (L0) {
        const float* vb = xv + (((size_t)b * TTs + t) * NNr + oct * 32) * DIN_;
        const float* mb = xm + (((size_t)b * TTs + t) * NNr + oct * 32) * DIN_;
        for (int i = tid; i < 32 * DIN_; i += 256) {
            int row = i / DIN_, d = i % DIN_;
            sh_xa[row >> 4][(row & 15) * K2 + d] =
                (_Float16)(vb[row * DIN_ + d] * mb[row * DIN_ + d]);
        }
    } else {
        const _Float16* hb = Xh + (((size_t)b * TTs + t) * NNr + oct * 32) * DIN_;
        for (int i = tid; i < 32 * DIN_; i += 256) {
            int row = i / DIN_, d = i % DIN_;
            sh_xa[row >> 4][(row & 15) * K2 + d] = hb[row * DIN_ + d];
        }
    }
    __syncthreads();

    // ---------- Phase 2: z = xa @ W + b, then gates ----------
    v16h xaf[KC];
#pragma unroll
    for (int kc = 0; kc < KC; ++kc) xaf[kc] = load_frag(sh_xa[s] + kc * 32, K2, lane);

    v8f za[4];
#pragma unroll
    for (int g = 0; g < 4; ++g)
        za[g] = splat8(bias[g * 64 + q * 16 + (lane & 15)]);

#pragma unroll
    for (int kc = 0; kc < KC; ++kc) {
#pragma unroll
        for (int g = 0; g < 4; ++g) {
            v16h bf = load_frag(WT + (size_t)(g * 64 + q * 16) * K2 + kc * 32, K2, lane);
            za[g] = __builtin_amdgcn_wmma_f32_16x16x32_f16(
                false, xaf[kc], false, bf, (short)0, za[g], false, false);
        }
    }

#pragma unroll
    for (int v = 0; v < 8; ++v) {
        int m  = r0 + v + ((lane >> 4) << 3);
        int hc = q * 16 + (lane & 15);
        float iv = za[0][v], fv = za[1][v], gv = za[2][v], ov = za[3][v];
        size_t ci = ((size_t)b * NNr + m) * HHh + hc;
        float cold = cbuf[ci];
        float cn = sigm(fv) * cold + sigm(iv) * tanhf(gv);
        float hn = sigm(ov) * tanhf(cn);
        cbuf[ci] = cn;
        hT_out[(size_t)b * HHh * NNr + hc * NNr + m] = (_Float16)hn;
        Hout[(((size_t)b * TTs + t) * NNr + m) * HHh + hc] = (_Float16)hn;
    }
}

// ---------------- output projection ----------------
// out[b,t,n,:] = H1[b,t,n,:] @ Wout + bout   (t < T-1). K=64, 32 cols.
// 256 threads / 8 waves; wave handles 16 rows of the flattened (b,t,n) space.
__global__ __launch_bounds__(256) void k_outproj(
    const _Float16* __restrict__ H1,   // [B,T,N,H] f16
    const _Float16* __restrict__ WoT,  // [32][64] j-major f16
    const float*    __restrict__ bout, // [32]
    float*          __restrict__ out)  // [B,T-1,N,32]
{
    const int tid = threadIdx.x, w = tid >> 5, lane = tid & 31;
    const int r0  = blockIdx.x * 128 + w * 16;   // tile-aligned: one (b,t) per tile
    const int koff = (lane >> 4) * 8;

    int rr = r0 + (lane & 15);
    int n = rr & 255, bt = rr >> 8;
    int t = bt % (TTs - 1), b = bt / (TTs - 1);
    const _Float16* ab = H1 + (((size_t)b * TTs + t) * NNr + n) * HHh;

    v16h af[2];
#pragma unroll
    for (int kc = 0; kc < 2; ++kc) {
        Frag16 f;
        const _Float16* p = ab + kc * 32 + koff;
        f.h[0] = *(const v8h*)p;
        f.h[1] = *(const v8h*)(p + 16);
        af[kc] = f.v;
    }

    v8f acc[2];
#pragma unroll
    for (int jt = 0; jt < 2; ++jt) acc[jt] = splat8(bout[jt * 16 + (lane & 15)]);

#pragma unroll
    for (int kc = 0; kc < 2; ++kc)
#pragma unroll
        for (int jt = 0; jt < 2; ++jt) {
            v16h bf = load_frag(WoT + jt * 16 * HHh + kc * 32, HHh, lane);
            acc[jt] = __builtin_amdgcn_wmma_f32_16x16x32_f16(
                false, af[kc], false, bf, (short)0, acc[jt], false, false);
        }

#pragma unroll
    for (int v = 0; v < 8; ++v) {
        int rm = r0 + v + ((lane >> 4) << 3);
        int n2 = rm & 255, bt2 = rm >> 8;
        int t2 = bt2 % (TTs - 1), b2 = bt2 / (TTs - 1);
        size_t o = (((size_t)b2 * (TTs - 1) + t2) * NNr + n2) * DIN0;
#pragma unroll
        for (int jt = 0; jt < 2; ++jt)
            out[o + jt * 16 + (lane & 15)] = acc[jt][v];
    }
}

// ---------------- host orchestration ----------------

extern "C" void kernel_launch(void* const* d_in, const int* in_sizes, int n_in,
                              void* d_out, int out_size, void* d_ws, size_t ws_size,
                              hipStream_t stream) {
    (void)in_sizes; (void)n_in; (void)out_size; (void)ws_size;
    const float* values = (const float*)d_in[0];
    const float* masks  = (const float*)d_in[1];
    const float* Amat   = (const float*)d_in[2];
    const float* W0     = (const float*)d_in[3];
    const float* b0     = (const float*)d_in[4];
    const float* W1     = (const float*)d_in[5];
    const float* b1     = (const float*)d_in[6];
    const float* Wout   = (const float*)d_in[7];
    const float* bout   = (const float*)d_in[8];
    float* out = (float*)d_out;

    // workspace carve-up (256B aligned)
    uintptr_t base = (uintptr_t)d_ws;
    size_t off = 0;
    auto carve = [&](size_t bytes) -> uintptr_t {
        uintptr_t p = base + off;
        off = (off + bytes + 255) & ~(size_t)255;
        return p;
    };
    _Float16* A16  = (_Float16*)carve((size_t)BB * NNr * NNr * 2);   // 1 MB
    _Float16* WT0  = (_Float16*)carve((size_t)256 * 96 * 2);
    _Float16* WT1  = (_Float16*)carve((size_t)256 * 128 * 2);
    _Float16* WoT  = (_Float16*)carve((size_t)32 * 64 * 2);
    _Float16* hT0  = (_Float16*)carve((size_t)BB * HHh * NNr * 2);
    _Float16* hT1  = (_Float16*)carve((size_t)BB * HHh * NNr * 2);
    float*    cbuf = (float*)   carve((size_t)BB * NNr * HHh * 4);
    _Float16* H0   = (_Float16*)carve((size_t)BB * TTs * NNr * HHh * 2); // 33.5 MB
    _Float16* H1   = (_Float16*)carve((size_t)BB * TTs * NNr * HHh * 2); // 33.5 MB

    // precompute: f16 operands in WMMA-friendly layouts
    k_f32_to_f16<<<2048, 256, 0, stream>>>(Amat, A16, BB * NNr * NNr);
    k_transpose_f16<<<96, 256, 0, stream>>>(W0, WT0, 96, 256);
    k_transpose_f16<<<128, 256, 0, stream>>>(W1, WT1, 128, 256);
    k_transpose_f16<<<8, 256, 0, stream>>>(Wout, WoT, 64, 32);

    const int nC = BB * NNr * HHh;      // 131072 f32
    const int nH = BB * HHh * NNr;      // 131072 f16

    // ---- layer 0 ----
    k_zero<<<512, 256, 0, stream>>>(cbuf, nC, hT0, nH);
    for (int t = 0; t < TTs; ++t) {
        const _Float16* hin = (t & 1) ? hT1 : hT0;
        _Float16*       hon = (t & 1) ? hT0 : hT1;
        k_step<DIN0, true><<<BB * 8, 256, 0, stream>>>(
            values, masks, (const _Float16*)nullptr,
            A16, WT0, b0, hin, hon, cbuf, H0, t);
    }
    // ---- layer 1 ----
    k_zero<<<512, 256, 0, stream>>>(cbuf, nC, hT0, nH);
    for (int t = 0; t < TTs; ++t) {
        const _Float16* hin = (t & 1) ? hT1 : hT0;
        _Float16*       hon = (t & 1) ? hT0 : hT1;
        k_step<HHh, false><<<BB * 8, 256, 0, stream>>>(
            (const float*)nullptr, (const float*)nullptr, H0,
            A16, WT1, b1, hin, hon, cbuf, H1, t);
    }
    // ---- output projection over B*(T-1)*N rows ----
    k_outproj<<<(BB * (TTs - 1) * NNr) / 128, 256, 0, stream>>>(H1, WoT, bout, out);
}
